// get_model_31610959298778
// MI455X (gfx1250) — matrix operand order repaired and checked by reading, not compiled, not run on hardware
//
#include <hip/hip_runtime.h>
#include <hip/hip_bf16.h>

// ---------------------------------------------------------------------------
// Swin window attention on gfx1250 (MI455X): one workgroup per window.
// All GEMMs use v_wmma_f32_16x16x32_bf16 (bf16 operands, f32 accumulate).
// Gather tables staged via GLOBAL_LOAD_ASYNC_TO_LDS (ASYNCcnt) when available.
// ---------------------------------------------------------------------------

#define DIMC   96
#define HEADS  6
#define HD     16
#define NTOK   64
#define TDIM   288     // 3*DIMC
#define TABLE  225

typedef __attribute__((ext_vector_type(16))) __bf16 v16bf;
typedef __attribute__((ext_vector_type(8)))  float  v8f;
typedef __attribute__((ext_vector_type(4)))  int    v4i;

union BF16Frag { v16bf v; __bf16 e[16]; };

#if __has_builtin(__builtin_amdgcn_global_load_async_to_lds_b128) && \
    __has_builtin(__builtin_amdgcn_global_load_async_to_lds_b32)  && \
    __has_builtin(__builtin_amdgcn_s_wait_asynccnt)
#define USE_ASYNC_LDS 1
#endif

// A fragment: 16x32 (MxK). element (m,k) at base[m*ms + k*ks]; k >= kmax -> 0.
__device__ inline v16bf load_a(const __bf16* base, int ms, int ks, int kmax) {
  const int lane = threadIdx.x & 31;
  const int m    = lane & 15;
  const int half = lane >> 4;
  BF16Frag f;
#pragma unroll
  for (int d = 0; d < 8; ++d) {
    int k = (d < 4) ? (half * 8 + 2 * d) : (16 + half * 8 + 2 * (d - 4));
    f.e[2 * d]     = (k     < kmax) ? base[m * ms + k * ks]       : (__bf16)0.0f;
    f.e[2 * d + 1] = (k + 1 < kmax) ? base[m * ms + (k + 1) * ks] : (__bf16)0.0f;
  }
  return f.v;
}

// B fragment: 32x16 (KxN). element (k,n) at base[k*ks + n*ns]; k >= kmax -> 0.
__device__ inline v16bf load_b(const __bf16* base, int ks, int ns, int kmax) {
  const int lane = threadIdx.x & 31;
  const int n    = lane & 15;
  const int half = lane >> 4;
  BF16Frag f;
#pragma unroll
  for (int d = 0; d < 8; ++d) {
    int k = (d < 4) ? (half * 8 + 2 * d) : (16 + half * 8 + 2 * (d - 4));
    f.e[2 * d]     = (k     < kmax) ? base[k * ks + n * ns]       : (__bf16)0.0f;
    f.e[2 * d + 1] = (k + 1 < kmax) ? base[(k + 1) * ks + n * ns] : (__bf16)0.0f;
  }
  return f.v;
}

__device__ inline v8f wmma_bf16(v16bf a, v16bf b, v8f c) {
  return __builtin_amdgcn_wmma_f32_16x16x32_bf16(
      false, a, false, b, (short)0, c, false, false);
}

// Vectorized f32 -> bf16 LDS staging (global_load_b128 + ds_store_b64).
__device__ inline void stage_bf16(__bf16* dst, const float* src, int count, int tid) {
  for (int i = tid * 4; i < count; i += 256 * 4) {
    float4 f = *(const float4*)(src + i);
    dst[i + 0] = (__bf16)f.x;
    dst[i + 1] = (__bf16)f.y;
    dst[i + 2] = (__bf16)f.z;
    dst[i + 3] = (__bf16)f.w;
  }
}

// LDS element counts
#define SM_X    (NTOK * DIMC)        // 6144  bf16
#define SM_W    (TDIM * DIMC)        // 27648 bf16
#define SM_PW   (DIMC * DIMC)        // 9216  bf16
#define SM_QKV  (NTOK * TDIM)        // 18432 bf16
#define SM_ATT  (HEADS * NTOK * NTOK)// 24576 bf16
#define SM_O    (NTOK * DIMC)        // 6144  bf16
#define SM_BF16 (SM_X + SM_W + SM_PW + SM_QKV + SM_ATT + SM_O)  // 92160
#define SM_RPB  1356                 // 225*6 padded
#define SM_F32  (TDIM + DIMC + SM_RPB)
#define SM_I32  (NTOK * NTOK)
#define SMEM_BYTES (SM_BF16 * 2 + SM_F32 * 4 + SM_I32 * 4)      // 207664

__global__ __launch_bounds__(256) void swin_window_attn(
    const float* __restrict__ x, const int* __restrict__ rpi,
    const float* __restrict__ qkv_w, const float* __restrict__ qkv_b,
    const float* __restrict__ proj_w, const float* __restrict__ proj_b,
    const float* __restrict__ rpb, float* __restrict__ out) {
  extern __shared__ char smem[];
  __bf16* sX    = (__bf16*)smem;
  __bf16* sW    = sX + SM_X;
  __bf16* sPW   = sW + SM_W;
  __bf16* sQKV  = sPW + SM_PW;
  __bf16* sAttn = sQKV + SM_QKV;
  __bf16* sO    = sAttn + SM_ATT;
  float*  sQB   = (float*)(sO + SM_O);
  float*  sPB   = sQB + TDIM;
  float*  sRpb  = sPB + DIMC;
  int*    sRpi  = (int*)(sRpb + SM_RPB);

  const int tid  = threadIdx.x;
  const int wave = tid >> 5;
  const int lane = tid & 31;
  const int bwin = blockIdx.x;

  // ---- Stage gather tables / biases (no conversion needed) ----
#ifdef USE_ASYNC_LDS
  {
    // rpi: 16384 B via b128 async copies (ASYNCcnt path)
    const char* g = (const char*)rpi;
    char*       l = (char*)sRpi;
    for (int i = tid * 16; i < SM_I32 * 4; i += 256 * 16)
      __builtin_amdgcn_global_load_async_to_lds_b128(
          (v4i*)(g + i), (v4i*)(l + i), 0, 0);
    for (int i = tid; i < TABLE * HEADS; i += 256)
      __builtin_amdgcn_global_load_async_to_lds_b32(
          (int*)(rpb + i), (int*)(sRpb + i), 0, 0);
    for (int i = tid; i < TDIM; i += 256)
      __builtin_amdgcn_global_load_async_to_lds_b32(
          (int*)(qkv_b + i), (int*)(sQB + i), 0, 0);
    for (int i = tid; i < DIMC; i += 256)
      __builtin_amdgcn_global_load_async_to_lds_b32(
          (int*)(proj_b + i), (int*)(sPB + i), 0, 0);
  }
#else
  for (int i = tid; i < SM_I32; i += 256) sRpi[i] = rpi[i];
  for (int i = tid; i < TABLE * HEADS; i += 256) sRpb[i] = rpb[i];
  for (int i = tid; i < TDIM; i += 256) sQB[i] = qkv_b[i];
  for (int i = tid; i < DIMC; i += 256) sPB[i] = proj_b[i];
#endif

  // ---- Stage x / weights with f32 -> bf16 downconvert (overlaps async) ----
  const float* xw = x + (size_t)bwin * NTOK * DIMC;
  stage_bf16(sX, xw, SM_X, tid);
  stage_bf16(sW, qkv_w, SM_W, tid);
  stage_bf16(sPW, proj_w, SM_PW, tid);

#ifdef USE_ASYNC_LDS
  __builtin_amdgcn_s_wait_asynccnt(0);
#endif
  __syncthreads();

  // ---- Phase A: QKV GEMM  (64x96) @ (96x288)^T : 4x18 tiles, 3 K-steps ----
  for (int t = wave * 9; t < wave * 9 + 9; ++t) {
    const int mt = t / 18, nt = t % 18;
    v8f acc = {};
#pragma unroll
    for (int kk = 0; kk < 3; ++kk) {
      v16bf a  = load_a(sX + mt * 16 * DIMC + kk * 32, DIMC, 1, 32);
      v16bf bm = load_b(sW + nt * 16 * DIMC + kk * 32, 1, DIMC, 32);  // W^T
      acc = wmma_bf16(a, bm, acc);
    }
    const int  n0   = nt * 16 + (lane & 15);
    const int  row0 = mt * 16 + ((lane >> 4) << 3);
    const float bv  = sQB[n0];
#pragma unroll
    for (int r = 0; r < 8; ++r)
      sQKV[(row0 + r) * TDIM + n0] = (__bf16)(acc[r] + bv);
  }
  __syncthreads();

  // ---- Phase B: per-head attention (wave h handles head h) ----
  if (wave < HEADS) {
    const int h = wave;
    const __bf16* Qb = sQKV + h * HD;             // (m,k) -> m*TDIM + k
    const __bf16* Kb = sQKV + DIMC + h * HD;      // B(k,n) = K[n][k]
    const __bf16* Vb = sQKV + 2 * DIMC + h * HD;  // B(k,n) = V[k][n]
    __bf16* sA = sAttn + h * NTOK * NTOK;
    const float scale = 0.25f;  // HD^-0.5
    const int half = lane >> 4;
    const int jcol = lane & 15;

    for (int ti = 0; ti < 4; ++ti) {
      v16bf aq = load_a(Qb + ti * 16 * TDIM, TDIM, 1, 16);   // K padded 16->32
      v8f acc[4];
#pragma unroll
      for (int tj = 0; tj < 4; ++tj) {
        v8f c = {};
        v16bf bk = load_b(Kb + tj * 16 * TDIM, 1, TDIM, 16);
        acc[tj] = wmma_bf16(aq, bk, c);
      }
      // row softmax: row i = ti*16 + r + 8*half, spread over 16 lanes x 4 tj
#pragma unroll
      for (int r = 0; r < 8; ++r) {
        const int i = ti * 16 + r + 8 * half;
        float vv[4];
        float mx = -1e30f;
#pragma unroll
        for (int tj = 0; tj < 4; ++tj) {
          const int j = tj * 16 + jcol;
          const float bsv = sRpb[sRpi[i * NTOK + j] * HEADS + h];
          vv[tj] = acc[tj][r] * scale + bsv;
          mx = fmaxf(mx, vv[tj]);
        }
#pragma unroll
        for (int s = 1; s < 16; s <<= 1) mx = fmaxf(mx, __shfl_xor(mx, s, 32));
        float sum = 0.0f;
#pragma unroll
        for (int tj = 0; tj < 4; ++tj) { vv[tj] = __expf(vv[tj] - mx); sum += vv[tj]; }
#pragma unroll
        for (int s = 1; s < 16; s <<= 1) sum += __shfl_xor(sum, s, 32);
        const float inv = __builtin_amdgcn_rcpf(sum);
#pragma unroll
        for (int tj = 0; tj < 4; ++tj)
          sA[i * NTOK + tj * 16 + jcol] = (__bf16)(vv[tj] * inv);
      }
    }

    // AV: (64x64) @ (64x16), 4 M-tiles x 2 K-steps
    for (int mt2 = 0; mt2 < 4; ++mt2) {
      v8f acc2 = {};
#pragma unroll
      for (int kk = 0; kk < 2; ++kk) {
        v16bf aa = load_a(sA + mt2 * 16 * NTOK + kk * 32, NTOK, 1, 32);
        v16bf bv = load_b(Vb + kk * 32 * TDIM, TDIM, 1, 32);
        acc2 = wmma_bf16(aa, bv, acc2);
      }
      const int n0   = lane & 15;
      const int row0 = mt2 * 16 + ((lane >> 4) << 3);
#pragma unroll
      for (int r = 0; r < 8; ++r)
        sO[(row0 + r) * DIMC + h * HD + n0] = (__bf16)acc2[r];
    }
  }
  __syncthreads();

  // ---- Phase C: projection (64x96) @ (96x96)^T + bias -> global f32 ----
  float* ow = out + (size_t)bwin * NTOK * DIMC;
  for (int t = wave * 3; t < wave * 3 + 3; ++t) {
    const int mt = t / 6, nt = t % 6;
    v8f acc = {};
#pragma unroll
    for (int kk = 0; kk < 3; ++kk) {
      v16bf a  = load_a(sO + mt * 16 * DIMC + kk * 32, DIMC, 1, 32);
      v16bf bm = load_b(sPW + nt * 16 * DIMC + kk * 32, 1, DIMC, 32);  // W^T
      acc = wmma_bf16(a, bm, acc);
    }
    const int  n0   = nt * 16 + (lane & 15);
    const int  row0 = mt * 16 + ((lane >> 4) << 3);
    const float bv  = sPB[n0];
#pragma unroll
    for (int r = 0; r < 8; ++r)
      ow[(row0 + r) * DIMC + n0] = acc[r] + bv;
  }
}

extern "C" void kernel_launch(void* const* d_in, const int* in_sizes, int n_in,
                              void* d_out, int out_size, void* d_ws, size_t ws_size,
                              hipStream_t stream) {
  (void)in_sizes; (void)n_in; (void)out_size; (void)d_ws; (void)ws_size;
  const float* x      = (const float*)d_in[0];
  const int*   rpi    = (const int*)d_in[1];
  const float* qkv_w  = (const float*)d_in[2];
  const float* qkv_b  = (const float*)d_in[3];
  const float* proj_w = (const float*)d_in[4];
  const float* proj_b = (const float*)d_in[5];
  const float* rpb    = (const float*)d_in[6];
  float* out = (float*)d_out;

  const int num_windows = 4096;
  swin_window_attn<<<num_windows, 256, SMEM_BYTES, stream>>>(
      x, rpi, qkv_w, qkv_b, proj_w, proj_b, rpb, out);
}
